// KLGCN_52106543235211
// MI455X (gfx1250) — matrix operand order repaired and checked by reading, not compiled, not run on hardware
//
#include <hip/hip_runtime.h>
#include <hip/hip_bf16.h>

typedef __attribute__((ext_vector_type(16))) _Float16 v16h;
typedef __attribute__((ext_vector_type(8)))  _Float16 v8h;
typedef __attribute__((ext_vector_type(8)))  float    v8f;

#define DIM  64
#define NNB  16
#define ROWS_PER_BLOCK 64
#define WAVES_PER_BLOCK 4
#define THREADS (WAVES_PER_BLOCK * 32)

__global__ __launch_bounds__(THREADS)
void klgcn_fused(const int* __restrict__ u, const int* __restrict__ v,
                 const int* __restrict__ user_neighbor, const int* __restrict__ item_neighbor,
                 const int* __restrict__ adj_ent, const int* __restrict__ adj_rel,
                 const float* __restrict__ usr_emb, const float* __restrict__ ent_emb,
                 const float* __restrict__ rel_emb, const float* __restrict__ agg_W,
                 const float* __restrict__ agg_b, float* __restrict__ out)
{
    __shared__ _Float16 sWt[DIM * DIM];                // agg_W TRANSPOSED as f16: sWt[n][k]
    __shared__ float    sB[DIM];                       // agg_b
    __shared__ _Float16 sComb[ROWS_PER_BLOCK * DIM];   // combined (f16, WMMA A input)
    __shared__ float    sUserF[ROWS_PER_BLOCK * DIM];  // user_final
    __shared__ float    sLiteI[ROWS_PER_BLOCK * DIM];  // lite_item mean

    const int tid = threadIdx.x;

    // ---- preload agg_W (transposed, f16) and agg_b into LDS ----
    #pragma unroll
    for (int i = 0; i < (DIM * DIM) / THREADS; ++i) {
        const int idx = tid + i * THREADS;       // idx = k*64 + n (row-major source)
        const int k = idx >> 6;
        const int n = idx & 63;
        sWt[n * DIM + k] = (_Float16)agg_W[idx]; // column-major in LDS
    }
    if (tid < DIM) sB[tid] = agg_b[tid];

    const int wave = tid >> 5;
    const int lane = tid & 31;
    const int R0   = wave * 16;                 // first row (block-local) of this wave
    const int blockRow0 = blockIdx.x * ROWS_PER_BLOCK;

    const int d0 = lane, d1 = lane + 32;        // 2 dims per lane (DIM=64, wave32)

    // ================= stage 1: gathers, attention, aggregation =================
    for (int r = 0; r < 16; ++r) {
        const int row = R0 + r;
        const int b   = blockRow0 + row;
        const int ub  = u[b];
        const int vb  = v[b];

        const float ue0 = usr_emb[ub * DIM + d0];
        const float ue1 = usr_emb[ub * DIM + d1];

        // attention scores: dot(user_emb, rel_emb[adj_rel[v,n]]) with wave32 reduction
        float sc[NNB];
        #pragma unroll
        for (int n = 0; n < NNB; ++n) {
            const int rid = adj_rel[vb * NNB + n];
            float p = ue0 * rel_emb[rid * DIM + d0] + ue1 * rel_emb[rid * DIM + d1];
            p += __shfl_xor(p, 1, 32);
            p += __shfl_xor(p, 2, 32);
            p += __shfl_xor(p, 4, 32);
            p += __shfl_xor(p, 8, 32);
            p += __shfl_xor(p, 16, 32);
            sc[n] = p;                          // broadcast to all lanes
        }
        // softmax over 16 neighbors (lane-uniform)
        float mx = sc[0];
        #pragma unroll
        for (int n = 1; n < NNB; ++n) mx = fmaxf(mx, sc[n]);
        float sum = 0.0f;
        #pragma unroll
        for (int n = 0; n < NNB; ++n) { sc[n] = __expf(sc[n] - mx); sum += sc[n]; }
        const float inv = 1.0f / sum;

        // attention-weighted neighbor aggregation
        float a0 = 0.0f, a1 = 0.0f;
        #pragma unroll
        for (int n = 0; n < NNB; ++n) {
            const int eid = adj_ent[vb * NNB + n];
            const float w = sc[n] * inv;
            a0 += w * ent_emb[eid * DIM + d0];
            a1 += w * ent_emb[eid * DIM + d1];
        }
        const float c0 = ent_emb[vb * DIM + d0] + a0;
        const float c1 = ent_emb[vb * DIM + d1] + a1;
        sComb[row * DIM + d0] = (_Float16)c0;
        sComb[row * DIM + d1] = (_Float16)c1;

        // lite_user / lite_item (mean of 16 gathered rows each)
        float lu0 = 0.f, lu1 = 0.f, li0 = 0.f, li1 = 0.f;
        #pragma unroll
        for (int n = 0; n < NNB; ++n) {
            const int iu = item_neighbor[b * NNB + n];
            const int ie = user_neighbor[b * NNB + n];
            lu0 += usr_emb[iu * DIM + d0];
            lu1 += usr_emb[iu * DIM + d1];
            li0 += ent_emb[ie * DIM + d0];
            li1 += ent_emb[ie * DIM + d1];
        }
        const float invn = 1.0f / NNB;
        sUserF[row * DIM + d0] = 0.5f * (lu0 * invn) + 0.5f * ue0;
        sUserF[row * DIM + d1] = 0.5f * (lu1 * invn) + 0.5f * ue1;
        sLiteI[row * DIM + d0] = li0 * invn;
        sLiteI[row * DIM + d1] = li1 * invn;
    }

    __syncthreads();

    // ================= stage 2: 16x64 = (16x64) @ (64x64) via WMMA =================
    const int q = lane & 15;   // position within half
    const int h = lane >> 4;   // lane half

    v8f acc[4];
    #pragma unroll
    for (int nt = 0; nt < 4; ++nt) {
        v8f z = {};
        acc[nt] = z;
    }

    #pragma unroll
    for (int t = 0; t < 2; ++t) {           // K-steps of 32
        // A tile (16x32 f16), row M = q:
        //   elems 0..7  -> K = 16h + e       (contiguous run of 8)
        //   elems 8..15 -> K = 16h + e       (contiguous run of 8)
        const _Float16* arow = &sComb[(R0 + q) * DIM + 32 * t + 16 * h];
        const v8h alo = *(const v8h*)(arow);        // 16B aligned
        const v8h ahi = *(const v8h*)(arow + 8);    // 16B aligned
        const v16h a = __builtin_shufflevector(alo, ahi,
            0, 1, 2, 3, 4, 5, 6, 7, 8, 9, 10, 11, 12, 13, 14, 15);

        #pragma unroll
        for (int nt = 0; nt < 4; ++nt) {    // N tiles of 16
            // B tile (32x16 f16), col N = 16nt+q: elem e -> K = 16h + e
            // sWt is [n][k], so the 16 elements are contiguous: one 32B load
            const v16h bb = *(const v16h*)&sWt[(16 * nt + q) * DIM + 32 * t + 16 * h];
            acc[nt] = __builtin_amdgcn_wmma_f32_16x16x32_f16(
                /*neg_a=*/false, a, /*neg_b=*/false, bb,
                /*c_mod=*/(short)0, acc[nt], /*reuse_a=*/false, /*reuse_b=*/false);
        }
    }

    // ================= epilogue: tanh, blend, dot, sigmoid =================
    // C/D layout: float j of acc -> row (j + 8h), column 16*nt + q
    float p[8];
    #pragma unroll
    for (int j = 0; j < 8; ++j) p[j] = 0.0f;

    #pragma unroll
    for (int nt = 0; nt < 4; ++nt) {
        const int col  = 16 * nt + q;
        const float bias = sB[col];
        #pragma unroll
        for (int j = 0; j < 8; ++j) {
            const int row = 8 * h + j;
            const float ite = tanhf(acc[nt][j] + bias);
            const float itf = 0.5f * sLiteI[(R0 + row) * DIM + col] + 0.5f * ite;
            p[j] += sUserF[(R0 + row) * DIM + col] * itf;
        }
    }
    // reduce over the 16 lanes of each half (columns of each row live there)
    #pragma unroll
    for (int j = 0; j < 8; ++j) {
        p[j] += __shfl_xor(p[j], 1, 32);
        p[j] += __shfl_xor(p[j], 2, 32);
        p[j] += __shfl_xor(p[j], 4, 32);
        p[j] += __shfl_xor(p[j], 8, 32);
    }
    if (q < 8) {
        const int row = 8 * h + q;
        const int b   = blockRow0 + R0 + row;
        out[b] = 1.0f / (1.0f + __expf(-p[q]));
    }
}

extern "C" void kernel_launch(void* const* d_in, const int* in_sizes, int n_in,
                              void* d_out, int out_size, void* d_ws, size_t ws_size,
                              hipStream_t stream) {
    (void)in_sizes; (void)n_in; (void)d_ws; (void)ws_size;
    const int*   u             = (const int*)d_in[0];
    const int*   v             = (const int*)d_in[1];
    const int*   user_neighbor = (const int*)d_in[2];
    const int*   item_neighbor = (const int*)d_in[3];
    const int*   adj_ent       = (const int*)d_in[4];
    const int*   adj_rel       = (const int*)d_in[5];
    const float* usr_emb       = (const float*)d_in[6];
    const float* ent_emb       = (const float*)d_in[7];
    const float* rel_emb       = (const float*)d_in[8];
    const float* agg_W         = (const float*)d_in[9];
    const float* agg_b         = (const float*)d_in[10];
    float* out = (float*)d_out;

    const int batch  = out_size;                         // 16384
    const int blocks = batch / ROWS_PER_BLOCK;           // 256
    klgcn_fused<<<blocks, THREADS, 0, stream>>>(
        u, v, user_neighbor, item_neighbor, adj_ent, adj_rel,
        usr_emb, ent_emb, rel_emb, agg_W, agg_b, out);
}
